// BaseGraphEncoder_60163901882838
// MI455X (gfx1250) — matrix operand order repaired and checked by reading, not compiled, not run on hardware
//
#include <hip/hip_runtime.h>
#include <hip/hip_bf16.h>

// MI455X / gfx1250 GIN graph encoder.
// - GEMMs: V_WMMA_F32_16X16X32_BF16, bf16 activations pre-converted once
// - B panels pre-packed in LDS image layout; staged with double-buffered
//   global_load_async_to_lds_b128 (ASYNCcnt) overlapping WMMA compute
// - Scatter/pool: f32 global atomics. All launches on `stream`.

typedef __attribute__((ext_vector_type(16))) __bf16 v16bf;
typedef __attribute__((ext_vector_type(8)))  float  v8f;

#define NNODES 100000
#define NEDGES 300000
#define DFEAT  256
#define NHIDC  512
#define NOUTC  768
#define NGRAPH 4096
#define BN_EPS 1e-5f

// Panel geometry: one (32K x 128N) B panel = 128 cols * 20 u32 (16 data + 4 pad)
#define PAN_U32   2560
#define PAN_CHNK  640          // 16B chunks per panel

__device__ __forceinline__ unsigned int f32_to_bf16_bits(float f) {
  unsigned int u = __float_as_uint(f);
  unsigned int r = u + 0x7FFFu + ((u >> 16) & 1u);   // round-to-nearest-even
  return r >> 16;
}
__device__ __forceinline__ unsigned int pack2_bf16(float lo, float hi) {
  return (f32_to_bf16_bits(hi) << 16) | f32_to_bf16_bits(lo);
}

union V16U8 { v16bf v; unsigned int u[8]; };

// ---------------------------------------------------------------------------
// WMMA GEMM:  C[M,NC] = epilogue(A[M,K] @ B[K,NC] + bias)
//   A: bf16 row-major.  B: panel-major image, each panel is the exact LDS blit:
//      idx = col*20 + (kpairLocal) , kpairLocal in [0,16), cols 0..127, 4 u32 pad.
//   MODE 0: relu -> bf16 out | MODE 1: relu -> eval-BN -> f32 out | MODE 2: f32 out
// Block: 256 threads = 8 waves; wave tile 16Mx64N; block tile 64Mx128N.
// ---------------------------------------------------------------------------
template<int K, int NC, int MODE>
__global__ __launch_bounds__(256)
void gemm_wmma(const unsigned short* __restrict__ Abf,
               const unsigned int* __restrict__ Bpan,
               const float* __restrict__ bias, void* __restrict__ Cp, int M,
               const float* __restrict__ bnG, const float* __restrict__ bnB,
               const float* __restrict__ bnM, const float* __restrict__ bnV) {
  __shared__ __attribute__((aligned(16))) unsigned int sB[2 * PAN_U32];

  const int tid  = threadIdx.x;
  const int lane = tid & 31;
  const int wave = tid >> 5;
  const int half = lane >> 4;          // 0: lanes 0-15, 1: lanes 16-31
  const int r    = lane & 15;
  const int mBase = blockIdx.x * 64 + (wave & 3) * 16;
  const int nBase = blockIdx.y * 128 + (wave >> 2) * 64;
  int row = mBase + r;
  if (row >= M) row = M - 1;           // clamp ragged M edge (stores predicated)

  // Issue one panel (3 uniform async b128 chunks per thread; clamped dups benign)
  auto issueB = [&](int buf, int kb) {
    const unsigned int* pb =
        Bpan + ((size_t)(kb >> 5) * (NC / 128) + blockIdx.y) * PAN_U32;
    const unsigned ldsBase = (unsigned)(buf * PAN_U32 * 4);
#pragma unroll
    for (int cc = 0; cc < 3; ++cc) {
      int c = tid + cc * 256;
      if (c >= PAN_CHNK) c = 0;
      const unsigned long long g = (unsigned long long)(pb + c * 4);
      const unsigned l = ldsBase + (unsigned)(c * 16);
      asm volatile("global_load_async_to_lds_b128 %0, %1, off"
                   :: "v"(l), "v"(g) : "memory");
    }
  };

  v8f acc[4] = {};

  issueB(0, 0);
  for (int kb = 0, it = 0; kb < K; kb += 32, ++it) {
    if (it) __syncthreads();           // buffer we are about to refill is free
    const int cur = it & 1;
    if (kb + 32 < K) {
      issueB(cur ^ 1, kb + 32);
      // async loads complete in order: <=3 outstanding => older batch landed
      asm volatile("s_wait_asynccnt 0x3" ::: "memory");
    } else {
      asm volatile("s_wait_asynccnt 0x0" ::: "memory");
    }
    __syncthreads();

    // ---- A tile 16x32 bf16: 2 x b128 per lane, lands in operand quads ----
    const unsigned int* Arow = (const unsigned int*)(Abf + (size_t)row * K + kb);
    const uint4 a0 = *(const uint4*)(Arow + half * 4);
    const uint4 a1 = *(const uint4*)(Arow + 8 + half * 4);
    V16U8 a;
    a.u[0] = a0.x; a.u[1] = a0.y; a.u[2] = a0.z; a.u[3] = a0.w;
    a.u[4] = a1.x; a.u[5] = a1.y; a.u[6] = a1.z; a.u[7] = a1.w;

    // ---- 4 B tiles 32x16 bf16: 2 x ds_load_b128 per tile per lane ----
    const unsigned int* sBc = sB + cur * PAN_U32;
#pragma unroll
    for (int t = 0; t < 4; ++t) {
      const int base = ((wave >> 2) * 64 + t * 16 + r) * 20 + half * 4;
      const uint4 b0 = *(const uint4*)(sBc + base);
      const uint4 b1 = *(const uint4*)(sBc + base + 8);
      V16U8 b;
      b.u[0] = b0.x; b.u[1] = b0.y; b.u[2] = b0.z; b.u[3] = b0.w;
      b.u[4] = b1.x; b.u[5] = b1.y; b.u[6] = b1.z; b.u[7] = b1.w;
      acc[t] = __builtin_amdgcn_wmma_f32_16x16x32_bf16(
          false, a.v, false, b.v, (short)0, acc[t], false, false);
    }
  }

  // ---- epilogue (C/D layout: VGPR j -> M = mBase + j + 8*half, lane%16 -> N)
  const bool full = (mBase + 16 <= M);   // wave-uniform fast path
#pragma unroll
  for (int t = 0; t < 4; ++t) {
    const int col = nBase + t * 16 + r;
    const float bv = bias[col];
    float ga = 0.f, bb = 0.f, mm = 0.f, iv = 0.f;
    if (MODE == 1) {
      ga = bnG[col]; bb = bnB[col]; mm = bnM[col];
      iv = rsqrtf(bnV[col] + BN_EPS);
    }
    auto emit = [&](int j) {
      const int m = mBase + j + half * 8;
      float v = acc[t][j] + bv;
      if (MODE == 0) {
        v = fmaxf(v, 0.0f);
        ((unsigned short*)Cp)[(size_t)m * NC + col] =
            (unsigned short)f32_to_bf16_bits(v);
      } else if (MODE == 1) {
        v = fmaxf(v, 0.0f);
        ((float*)Cp)[(size_t)m * NC + col] = (v - mm) * (ga * iv) + bb;
      } else {
        ((float*)Cp)[(size_t)m * NC + col] = v;
      }
    };
    if (full) {
#pragma unroll
      for (int j = 0; j < 8; ++j) emit(j);
    } else {
#pragma unroll
      for (int j = 0; j < 8; ++j)
        if (mBase + j + half * 8 < M) emit(j);
    }
  }
}

// hin = (1 + eps) * h      (float4 vectorized)
__global__ void init_hin(const float* __restrict__ h, const float* __restrict__ epsp,
                         float* __restrict__ hin, int n4) {
  int i = blockIdx.x * blockDim.x + threadIdx.x;
  if (i >= n4) return;
  const float s = 1.0f + epsp[0];
  float4 v = ((const float4*)h)[i];
  v.x *= s; v.y *= s; v.z *= s; v.w *= s;
  ((float4*)hin)[i] = v;
}

// hin[dst] += h[src]  over edges; 64 threads/edge, 4 floats/thread
__global__ void scatter_add(const float* __restrict__ h, const int* __restrict__ ei,
                            float* __restrict__ hin) {
  long long tid = (long long)blockIdx.x * blockDim.x + threadIdx.x;
  long long e = tid >> 6;
  if (e >= NEDGES) return;
  const int d4 = (int)(tid & 63) << 2;
  const int s = ei[e];
  const int t = ei[NEDGES + e];
  const float4 v = *(const float4*)(h + (size_t)s * DFEAT + d4);
  float* o = hin + (size_t)t * DFEAT + d4;
  atomicAdd(o + 0, v.x); atomicAdd(o + 1, v.y);
  atomicAdd(o + 2, v.z); atomicAdd(o + 3, v.w);
}

// f32 -> bf16 (packed pairs, row-major preserved); n2 = elements/2
__global__ void cvt_f32_bf16(const float* __restrict__ in,
                             unsigned int* __restrict__ out, int n2) {
  int i = blockIdx.x * blockDim.x + threadIdx.x;
  if (i >= n2) return;
  const float2 f = ((const float2*)in)[i];
  out[i] = pack2_bf16(f.x, f.y);
}

// Pack W[K x NC] f32 -> panel-major bf16 image matching the LDS blit layout.
// panel p = (kb/32)*(NC/128) + (nBlk/128); within panel: idx = col*20 + s,
// s in [0,16): kpair row = (kb/2)+s, col = nBlk+col ; s in [16,20): zero pad.
__global__ void pack_panels(const float* __restrict__ W,
                            unsigned int* __restrict__ out, int K, int Nc) {
  int i = blockIdx.x * blockDim.x + threadIdx.x;
  const int panelsN = Nc / 128;
  const int total = (K / 32) * panelsN * PAN_U32;
  if (i >= total) return;
  const int p = i / PAN_U32, q = i - p * PAN_U32;
  const int col = q / 20, s = q - col * 20;
  const int pk = p / panelsN, pn = p - pk * panelsN;
  unsigned v = 0;
  if (s < 16) {
    const int kpair = pk * 16 + s;
    const int n = pn * 128 + col;
    v = pack2_bf16(W[(size_t)(2 * kpair) * Nc + n],
                   W[(size_t)(2 * kpair + 1) * Nc + n]);
  }
  out[i] = v;
}

__global__ void zero_pool(float* pooled, float* cnt) {
  int i = blockIdx.x * blockDim.x + threadIdx.x;
  if (i < NGRAPH * DFEAT) pooled[i] = 0.0f;
  else if (i < NGRAPH * DFEAT + NGRAPH) cnt[i - NGRAPH * DFEAT] = 0.0f;
}

__global__ void pool_acc(const float* __restrict__ h, const int* __restrict__ batch,
                         float* __restrict__ pooled, float* __restrict__ cnt) {
  long long tid = (long long)blockIdx.x * blockDim.x + threadIdx.x;
  long long n = tid >> 6;
  if (n >= NNODES) return;
  const int d4 = (int)(tid & 63) << 2;
  const int g = batch[n];
  const float4 v = *(const float4*)(h + (size_t)n * DFEAT + d4);
  float* o = pooled + (size_t)g * DFEAT + d4;
  atomicAdd(o + 0, v.x); atomicAdd(o + 1, v.y);
  atomicAdd(o + 2, v.z); atomicAdd(o + 3, v.w);
  if (d4 == 0) atomicAdd(cnt + g, 1.0f);
}

__global__ void pool_div(float* __restrict__ pooled, const float* __restrict__ cnt) {
  int i = blockIdx.x * blockDim.x + threadIdx.x;
  if (i >= NGRAPH * DFEAT) return;
  pooled[i] /= fmaxf(cnt[i >> 8], 1.0f);   // DFEAT == 256
}

extern "C" void kernel_launch(void* const* d_in, const int* in_sizes, int n_in,
                              void* d_out, int out_size, void* d_ws, size_t ws_size,
                              hipStream_t stream) {
  const float* x    = (const float*)d_in[0];
  const int*   ei   = (const int*)  d_in[1];
  const int*   bat  = (const int*)  d_in[2];
  const float* gw1  = (const float*)d_in[3];
  const float* gb1  = (const float*)d_in[4];
  const float* gw2  = (const float*)d_in[5];
  const float* gb2  = (const float*)d_in[6];
  const float* geps = (const float*)d_in[7];
  const float* bng  = (const float*)d_in[8];
  const float* bnb  = (const float*)d_in[9];
  const float* bnm  = (const float*)d_in[10];
  const float* bnv  = (const float*)d_in[11];
  const float* wp1  = (const float*)d_in[12];
  const float* bp1  = (const float*)d_in[13];
  const float* wp2  = (const float*)d_in[14];
  const float* bp2  = (const float*)d_in[15];

  char* ws = (char*)d_ws;
  size_t off = 0;
  auto take = [&](size_t bytes) -> char* {
    char* p = ws + off;
    off += (bytes + 255) & ~(size_t)255;
    return p;
  };
  // panel-image u32 counts
  const int W1IMG = (DFEAT / 32) * (NHIDC / 128) * PAN_U32;   // 81920
  const int W2IMG = (NHIDC / 32) * (DFEAT / 128) * PAN_U32;   // 81920
  const int P1IMG = (DFEAT / 32) * (NHIDC / 128) * PAN_U32;   // 81920
  const int P2IMG = (NHIDC / 32) * (NOUTC / 128) * PAN_U32;   // 245760

  float*          hin   = (float*)         take((size_t)NNODES * DFEAT * 4);
  float*          hbuf  = (float*)         take((size_t)NNODES * DFEAT * 4);
  unsigned short* hbf   = (unsigned short*)take((size_t)NNODES * DFEAT * 2);
  unsigned short* c1    = (unsigned short*)take((size_t)NNODES * NHIDC * 2);
  unsigned int*   w1p   = (unsigned int*)  take((size_t)4 * W1IMG * 4);
  unsigned int*   w2p   = (unsigned int*)  take((size_t)4 * W2IMG * 4);
  unsigned int*   wp1p  = (unsigned int*)  take((size_t)P1IMG * 4);
  unsigned int*   wp2p  = (unsigned int*)  take((size_t)P2IMG * 4);
  float*          pool  = (float*)         take((size_t)NGRAPH * DFEAT * 4);
  float*          cnt   = (float*)         take((size_t)NGRAPH * 4);
  unsigned short* poolb = (unsigned short*)take((size_t)NGRAPH * DFEAT * 2);
  unsigned short* p1    = (unsigned short*)take((size_t)NGRAPH * NHIDC * 2);

  const int PT = 256;
  // --- pack all weights to panel-major bf16 images (tiny, once per call) ---
  for (int l = 0; l < 4; ++l) {
    pack_panels<<<(W1IMG + PT - 1) / PT, PT, 0, stream>>>(
        gw1 + (size_t)l * DFEAT * NHIDC, w1p + (size_t)l * W1IMG, DFEAT, NHIDC);
    pack_panels<<<(W2IMG + PT - 1) / PT, PT, 0, stream>>>(
        gw2 + (size_t)l * NHIDC * DFEAT, w2p + (size_t)l * W2IMG, NHIDC, DFEAT);
  }
  pack_panels<<<(P1IMG + PT - 1) / PT, PT, 0, stream>>>(wp1, wp1p, DFEAT, NHIDC);
  pack_panels<<<(P2IMG + PT - 1) / PT, PT, 0, stream>>>(wp2, wp2p, NHIDC, NOUTC);

  // --- 4 GIN layers ---
  const int bnidx[4] = {0, 0, 1, 2};  // reference keeps the bn-index bug
  const float* hprev = x;
  const int n4 = NNODES * DFEAT / 4;
  const int n2 = NNODES * DFEAT / 2;
  const long long sth = (long long)NEDGES * 64;
  for (int l = 0; l < 4; ++l) {
    init_hin<<<(n4 + PT - 1) / PT, PT, 0, stream>>>(hprev, geps + l, hin, n4);
    scatter_add<<<(unsigned)((sth + PT - 1) / PT), PT, 0, stream>>>(hprev, ei, hin);
    cvt_f32_bf16<<<(n2 + PT - 1) / PT, PT, 0, stream>>>(hin, (unsigned int*)hbf, n2);

    dim3 g1((NNODES + 63) / 64, NHIDC / 128);
    gemm_wmma<DFEAT, NHIDC, 0><<<g1, 256, 0, stream>>>(
        hbf, w1p + (size_t)l * W1IMG, gb1 + (size_t)l * NHIDC, c1, NNODES,
        nullptr, nullptr, nullptr, nullptr);

    dim3 g2((NNODES + 63) / 64, DFEAT / 128);
    const int bi = bnidx[l];
    gemm_wmma<NHIDC, DFEAT, 1><<<g2, 256, 0, stream>>>(
        c1, w2p + (size_t)l * W2IMG, gb2 + (size_t)l * DFEAT, hbuf, NNODES,
        bng + (size_t)bi * DFEAT, bnb + (size_t)bi * DFEAT,
        bnm + (size_t)bi * DFEAT, bnv + (size_t)bi * DFEAT);
    hprev = hbuf;
  }

  // --- global mean pool ---
  const int zt = NGRAPH * DFEAT + NGRAPH;
  zero_pool<<<(zt + PT - 1) / PT, PT, 0, stream>>>(pool, cnt);
  const long long pth = (long long)NNODES * 64;
  pool_acc<<<(unsigned)((pth + PT - 1) / PT), PT, 0, stream>>>(hprev, bat, pool, cnt);
  pool_div<<<(NGRAPH * DFEAT + PT - 1) / PT, PT, 0, stream>>>(pool, cnt);
  const int g2n = NGRAPH * DFEAT / 2;
  cvt_f32_bf16<<<(g2n + PT - 1) / PT, PT, 0, stream>>>(pool, (unsigned int*)poolb, g2n);

  // --- MLP head ---
  dim3 gh1((NGRAPH + 63) / 64, NHIDC / 128);
  gemm_wmma<DFEAT, NHIDC, 0><<<gh1, 256, 0, stream>>>(
      poolb, wp1p, bp1, p1, NGRAPH, nullptr, nullptr, nullptr, nullptr);
  dim3 gh2((NGRAPH + 63) / 64, NOUTC / 128);
  gemm_wmma<NHIDC, NOUTC, 2><<<gh2, 256, 0, stream>>>(
      p1, wp2p, bp2, (float*)d_out, NGRAPH, nullptr, nullptr, nullptr, nullptr);
}
